// Model_74268574483003
// MI455X (gfx1250) — matrix operand order repaired and checked by reading, not compiled, not run on hardware
//
#include <hip/hip_runtime.h>
#include <math.h>

// ---------------- types ----------------
typedef __bf16 bf16_t;
typedef bf16_t v16bf __attribute__((ext_vector_type(16)));
typedef bf16_t v8bf  __attribute__((ext_vector_type(8)));
typedef float  v8f   __attribute__((ext_vector_type(8)));

// ---------------- model constants ----------------
#define LAYERS 8
#define DMODEL 1072
#define KPAD   1088   // 1072 padded to mult of 32
#define NHEAD  16
#define DHEAD  67
#define DQPAD  96     // 67 padded to mult of 32 (K side of Q.K^T)
#define DVPAD  80     // 67 padded to mult of 16 (N side of attn.V)
#define SEQ    2048
#define BATCH  2
#define NCLS   300
#define NCLSP  304
#define D3     3216   // 3*D
#define D4     4288   // 4*D (already mult of 32)
#define RWIN   32
#define INV_SCALE 0.1221694322f   // 1/sqrt(67)

// ---------------- WMMA helpers ----------------
__device__ __forceinline__ v8f wmma_bf16(v16bf a, v16bf b, v8f c) {
  return __builtin_amdgcn_wmma_f32_16x16x32_bf16(false, a, false, b, (short)0, c,
                                                 false, false);
}

// A fragment (16x32, row-major source, ld = padded K):
// lane<16: row=lane, K = kb+{0..7, 16..23}; lane>=16: row=lane-16, K = kb+{8..15, 24..31}
__device__ __forceinline__ v16bf load_a_frag(const bf16_t* __restrict__ base, int ld,
                                             int row, int kb, int lane) {
  const bf16_t* p = base + (size_t)row * ld + kb + ((lane < 16) ? 0 : 8);
  v16bf a;
  ((v8bf*)&a)[0] = *(const v8bf*)(p);
  ((v8bf*)&a)[1] = *(const v8bf*)(p + 16);
  return a;
}

// B fragment (32x16) taken from W stored as N x K row-major (computing A.W^T):
// lane<16: col=lane, K = kb+{0..7, 16..23}; lane>=16: col=lane-16, K = kb+{8..15, 24..31}
__device__ __forceinline__ v16bf load_b_frag(const bf16_t* __restrict__ base, int ld,
                                             int nrow, int kb, int lane) {
  const bf16_t* p = base + (size_t)nrow * ld + kb + ((lane < 16) ? 0 : 8);
  v16bf b;
  ((v8bf*)&b)[0] = *(const v8bf*)(p);
  ((v8bf*)&b)[1] = *(const v8bf*)(p + 16);
  return b;
}

// ---------------- elementwise kernels ----------------
__global__ void embed_kernel(const int* __restrict__ ids, const float* __restrict__ emb,
                             float* __restrict__ x, int total) {
  int i = blockIdx.x * blockDim.x + threadIdx.x;
  if (i >= total) return;
  int d = i % DMODEL;
  int t = i / DMODEL;
  x[i] = emb[(size_t)ids[t] * DMODEL + d];
}

__global__ void convert_pad_kernel(const float* __restrict__ W, bf16_t* __restrict__ Wb,
                                   int Nr, int Kr, int Np, int Kp) {
  int i = blockIdx.x * blockDim.x + threadIdx.x;
  if (i >= Np * Kp) return;
  int k = i % Kp, n = i / Kp;
  float v = (n < Nr && k < Kr) ? W[(size_t)n * Kr + k] : 0.0f;
  Wb[i] = (bf16_t)v;
}

// LayerNorm of one row -> bf16 padded row (pads to Kp with zeros)
__global__ void layernorm_bf16(const float* __restrict__ xin, const float* __restrict__ w,
                               const float* __restrict__ bvec, bf16_t* __restrict__ out,
                               int rs_mul, int rs_add) {
  __shared__ float red[8];
  __shared__ float stat[2];
  int r = blockIdx.x;
  const float* xr = xin + (size_t)(r * rs_mul + rs_add) * DMODEL;
  int tid = threadIdx.x, lane = tid & 31, wave = tid >> 5;

  float s = 0.f;
  for (int d = tid; d < DMODEL; d += blockDim.x) s += xr[d];
  for (int o = 16; o > 0; o >>= 1) s += __shfl_xor(s, o, 32);
  if (lane == 0) red[wave] = s;
  __syncthreads();
  if (tid == 0) {
    float t = 0.f;
    for (int wv = 0; wv < 8; wv++) t += red[wv];
    stat[0] = t / (float)DMODEL;
  }
  __syncthreads();
  float mean = stat[0];

  float v = 0.f;
  for (int d = tid; d < DMODEL; d += blockDim.x) {
    float t = xr[d] - mean; v += t * t;
  }
  for (int o = 16; o > 0; o >>= 1) v += __shfl_xor(v, o, 32);
  if (lane == 0) red[wave] = v;
  __syncthreads();
  if (tid == 0) {
    float t = 0.f;
    for (int wv = 0; wv < 8; wv++) t += red[wv];
    stat[1] = rsqrtf(t / (float)DMODEL + 1e-6f);
  }
  __syncthreads();
  float rstd = stat[1];

  for (int d = tid; d < KPAD; d += blockDim.x) {
    float o = (d < DMODEL) ? ((xr[d] - mean) * rstd * w[d] + bvec[d]) : 0.f;
    out[(size_t)r * KPAD + d] = (bf16_t)o;
  }
}

// qkv (B,S,3D) bf16 -> q/k (B,H,S,96) padded, vT (B,H,80,S) padded+transposed
__global__ void repack_qkv_kernel(const bf16_t* __restrict__ qkv, bf16_t* __restrict__ qB,
                                  bf16_t* __restrict__ kB, bf16_t* __restrict__ vT) {
  int i = blockIdx.x * blockDim.x + threadIdx.x;
  const int total = BATCH * NHEAD * SEQ * DQPAD;
  if (i >= total) return;
  int dp = i % DQPAD;
  int s  = (i / DQPAD) % SEQ;
  int h  = (i / (DQPAD * SEQ)) % NHEAD;
  int b  = i / (DQPAD * SEQ * NHEAD);
  size_t row = ((size_t)(b * SEQ + s)) * D3 + h * (3 * DHEAD);
  size_t qi  = (((size_t)(b * NHEAD + h)) * SEQ + s) * DQPAD + dp;
  bf16_t z = (bf16_t)0.0f;
  qB[qi] = (dp < DHEAD) ? qkv[row + dp] : z;
  kB[qi] = (dp < DHEAD) ? qkv[row + DHEAD + dp] : z;
  if (dp < DVPAD) {
    size_t vi = (((size_t)(b * NHEAD + h)) * DVPAD + dp) * SEQ + s;
    vT[vi] = (dp < DHEAD) ? qkv[row + 2 * DHEAD + dp] : z;
  }
}

__global__ void mask_logits_kernel(const float* __restrict__ logits,
                                   const int* __restrict__ mask, float* __restrict__ out) {
  int i = blockIdx.x * blockDim.x + threadIdx.x;
  if (i >= BATCH * NCLS) return;
  int b = i / NCLS, c = i % NCLS;
  float v = logits[b * NCLSP + c];
  out[i] = (mask[i] == 0) ? -__builtin_inff() : v;
}

// ---------------- WMMA GEMM: out = A(M x Kp) . W(N x Kp)^T + bias ----------------
// mode bit0: bf16 output; bit1: exact GELU epilogue; bit2: residual += (f32 out)
__global__ __launch_bounds__(128)
void gemm_bf16_wmma(const bf16_t* __restrict__ A, const bf16_t* __restrict__ W,
                    const float* __restrict__ bias, float* __restrict__ Cf,
                    bf16_t* __restrict__ Cb, int Kp, int N_real, int M_real,
                    int Npad, int ldc, int mode) {
  int lane = threadIdx.x & 31;
  int wave = threadIdx.x >> 5;
  int wm = wave >> 1, wn = wave & 1;
  int mbase = blockIdx.y * 64 + wm * 32;
  int nbase = blockIdx.x * 128 + wn * 64;

  v8f acc[2][4];
  for (int i = 0; i < 2; i++)
    for (int j = 0; j < 4; j++)
      for (int r = 0; r < 8; r++) acc[i][j][r] = 0.f;

  for (int kb = 0; kb < Kp; kb += 32) {
    v16bf a0 = load_a_frag(A, Kp, mbase + (lane & 15), kb, lane);
    v16bf a1 = load_a_frag(A, Kp, mbase + 16 + (lane & 15), kb, lane);
    v16bf bfr[4];
#pragma unroll
    for (int j = 0; j < 4; j++) {
      int nrow = nbase + j * 16 + (lane & 15);
      if (nrow >= Npad) nrow = Npad - 1;   // safe clamp; stores are masked
      bfr[j] = load_b_frag(W, Kp, nrow, kb, lane);
    }
#pragma unroll
    for (int j = 0; j < 4; j++) {
      acc[0][j] = wmma_bf16(a0, bfr[j], acc[0][j]);
      acc[1][j] = wmma_bf16(a1, bfr[j], acc[1][j]);
    }
  }

#pragma unroll
  for (int i = 0; i < 2; i++)
#pragma unroll
    for (int j = 0; j < 4; j++) {
      int n = nbase + j * 16 + (lane & 15);
      if (n >= N_real) continue;
      float bi = bias[n];
#pragma unroll
      for (int r = 0; r < 8; r++) {
        int m = mbase + i * 16 + r + ((lane < 16) ? 0 : 8);
        if (m >= M_real) continue;
        float v = acc[i][j][r] + bi;
        if (mode & 2) v = 0.5f * v * (1.0f + erff(v * 0.70710678118f));
        size_t ci = (size_t)m * ldc + n;
        if (mode & 1)      Cb[ci] = (bf16_t)v;
        else if (mode & 4) Cf[ci] += v;
        else               Cf[ci] = v;
      }
    }
}

// ---------------- fused attention ----------------
// One workgroup per (b, h, 16-row i-tile). LDS: scores f32[16][S], attn bf16[16][S],
// pos f32[16][80], rowsum f32[16]. pos only needs cols 0..64 (clip window).
#define SMEM_SCORES (16 * SEQ * 4)
#define SMEM_ATTN   (16 * SEQ * 2)
#define SMEM_POS    (16 * DVPAD * 4)
#define SMEM_TOTAL  (SMEM_SCORES + SMEM_ATTN + SMEM_POS + 16 * 4)

__global__ __launch_bounds__(256)
void attention_kernel(const bf16_t* __restrict__ qB, const bf16_t* __restrict__ kB,
                      const bf16_t* __restrict__ vT, const bf16_t* __restrict__ Wrelb,
                      const float* __restrict__ brel, float* __restrict__ x) {
  extern __shared__ char smem_raw[];
  float*  scores = (float*)smem_raw;
  bf16_t* attnb  = (bf16_t*)(smem_raw + SMEM_SCORES);
  float*  posb   = (float*)(smem_raw + SMEM_SCORES + SMEM_ATTN);
  float*  rowsum = (float*)(smem_raw + SMEM_SCORES + SMEM_ATTN + SMEM_POS);

  int wg = blockIdx.x;
  int it = wg & 127;          // S/16 tiles
  int h  = (wg >> 7) & (NHEAD - 1);
  int b  = wg >> 11;
  int lane = threadIdx.x & 31;
  int wave = threadIdx.x >> 5;

  const bf16_t* qbase = qB + ((size_t)(b * NHEAD + h)) * SEQ * DQPAD;
  const bf16_t* kbase = kB + ((size_t)(b * NHEAD + h)) * SEQ * DQPAD;
  const bf16_t* vbase = vT + ((size_t)(b * NHEAD + h)) * DVPAD * SEQ;

  // Q fragments for this 16-row tile (K = 96 -> 3 fragments), reused everywhere
  v16bf qf[3];
#pragma unroll
  for (int t = 0; t < 3; t++)
    qf[t] = load_a_frag(qbase, DQPAD, it * 16 + (lane & 15), t * 32, lane);

  // wave 0: relative-position bias rp = q . Wrel^T + brel  (only first 65 cols used)
  if (wave == 0) {
#pragma unroll
    for (int nt = 0; nt < 5; nt++) {
      v8f acc; for (int r = 0; r < 8; r++) acc[r] = 0.f;
#pragma unroll
      for (int t = 0; t < 3; t++) {
        v16bf bw = load_b_frag(Wrelb, DQPAD, nt * 16 + (lane & 15), t * 32, lane);
        acc = wmma_bf16(qf[t], bw, acc);
      }
      int o = nt * 16 + (lane & 15);
      float bb = brel[o];
#pragma unroll
      for (int r = 0; r < 8; r++) {
        int m = r + ((lane < 16) ? 0 : 8);
        posb[m * DVPAD + o] = acc[r] + bb;
      }
    }
  }

  // content scores: each wave sweeps j-tiles strided by 8
  for (int jt = wave; jt < SEQ / 16; jt += 8) {
    v8f acc; for (int r = 0; r < 8; r++) acc[r] = 0.f;
#pragma unroll
    for (int t = 0; t < 3; t++) {
      v16bf kf = load_b_frag(kbase, DQPAD, jt * 16 + (lane & 15), t * 32, lane);
      acc = wmma_bf16(qf[t], kf, acc);
    }
    int j = jt * 16 + (lane & 15);
#pragma unroll
    for (int r = 0; r < 8; r++) {
      int m = r + ((lane < 16) ? 0 : 8);
      scores[m * SEQ + j] = acc[r] * INV_SCALE;
    }
  }
  __syncthreads();

  // softmax: wave w handles rows 2w, 2w+1; store unnormalized exp as bf16
  for (int rr = 0; rr < 2; rr++) {
    int m  = wave * 2 + rr;
    int ig = it * 16 + m;
    float mx = -3.4e38f;
    for (int j = lane; j < SEQ; j += 32) {
      int off = j - ig; off = off < -RWIN ? -RWIN : (off > RWIN ? RWIN : off);
      float v = scores[m * SEQ + j] + posb[m * DVPAD + off + RWIN];
      mx = fmaxf(mx, v);
    }
    for (int o = 16; o > 0; o >>= 1) mx = fmaxf(mx, __shfl_xor(mx, o, 32));
    float sum = 0.f;
    for (int j = lane; j < SEQ; j += 32) {
      int off = j - ig; off = off < -RWIN ? -RWIN : (off > RWIN ? RWIN : off);
      float v = scores[m * SEQ + j] + posb[m * DVPAD + off + RWIN];
      float e = __expf(v - mx);
      attnb[m * SEQ + j] = (bf16_t)e;
      sum += e;
    }
    for (int o = 16; o > 0; o >>= 1) sum += __shfl_xor(sum, o, 32);
    if (lane == 0) rowsum[m] = sum;
  }
  __syncthreads();

  // attn @ V: waves 0..4 each own one 16-col tile of the 80-wide padded output
  if (wave < 5) {
    int nt = wave;
    v8f acc; for (int r = 0; r < 8; r++) acc[r] = 0.f;
    for (int kb = 0; kb < SEQ; kb += 32) {
      v16bf af;
      {
        int m = lane & 15;
        const bf16_t* p = attnb + m * SEQ + kb + ((lane < 16) ? 0 : 8);
        ((v8bf*)&af)[0] = *(const v8bf*)(p);
        ((v8bf*)&af)[1] = *(const v8bf*)(p + 16);
      }
      v16bf vf = load_b_frag(vbase, SEQ, nt * 16 + (lane & 15), kb, lane);
      acc = wmma_bf16(af, vf, acc);
    }
    int d = nt * 16 + (lane & 15);
    if (d < DHEAD) {
#pragma unroll
      for (int r = 0; r < 8; r++) {
        int m  = r + ((lane < 16) ? 0 : 8);
        int ig = it * 16 + m;
        float o = acc[r] / rowsum[m];
        x[((size_t)(b * SEQ + ig)) * DMODEL + h * DHEAD + d] += o;   // disjoint writes
      }
    }
  }
}

// ---------------- host orchestration ----------------
extern "C" void kernel_launch(void* const* d_in, const int* in_sizes, int n_in,
                              void* d_out, int out_size, void* d_ws, size_t ws_size,
                              hipStream_t stream) {
  const int*   ids   = (const int*)d_in[0];
  const int*   mask  = (const int*)d_in[1];
  const float* emb   = (const float*)d_in[2];
  const float* ln1_w = (const float*)d_in[3];
  const float* ln1_b = (const float*)d_in[4];
  const float* Wqkv  = (const float*)d_in[5];
  const float* bqkv  = (const float*)d_in[6];
  const float* Wrel  = (const float*)d_in[7];
  const float* brel  = (const float*)d_in[8];
  const float* ln2_w = (const float*)d_in[9];
  const float* ln2_b = (const float*)d_in[10];
  const float* W1    = (const float*)d_in[11];
  const float* b1    = (const float*)d_in[12];
  const float* W2    = (const float*)d_in[13];
  const float* b2    = (const float*)d_in[14];
  const float* lnf_w = (const float*)d_in[15];
  const float* lnf_b = (const float*)d_in[16];
  const float* Wh    = (const float*)d_in[17];
  const float* bh    = (const float*)d_in[18];
  float* out = (float*)d_out;

  // workspace carve (bytes, 256-aligned)
  char* ws = (char*)d_ws;
  size_t off = 0;
  auto carve = [&](size_t bytes) -> char* {
    char* p = ws + off;
    off = (off + bytes + 255) & ~(size_t)255;
    return p;
  };
  const int M = BATCH * SEQ;                                   // 4096
  float*  x_f    = (float*)carve((size_t)M * DMODEL * 4);      // activations
  bf16_t* hb     = (bf16_t*)carve((size_t)M * KPAD * 2);       // LN output (padded K)
  bf16_t* qkvb   = (bf16_t*)carve((size_t)M * D3 * 2);         // qkv bf16
  bf16_t* qBb    = (bf16_t*)carve((size_t)BATCH * NHEAD * SEQ * DQPAD * 2);
  bf16_t* kBb    = (bf16_t*)carve((size_t)BATCH * NHEAD * SEQ * DQPAD * 2);
  bf16_t* vTb    = (bf16_t*)carve((size_t)BATCH * NHEAD * DVPAD * SEQ * 2);
  bf16_t* h1b    = (bf16_t*)carve((size_t)M * D4 * 2);         // FFN hidden (bf16, GELU'd)
  bf16_t* Wb     = (bf16_t*)carve((size_t)D4 * KPAD * 2);      // reusable weight scratch
  bf16_t* Wrelb  = (bf16_t*)carve((size_t)DVPAD * DQPAD * 2);
  bf16_t* headA  = (bf16_t*)carve((size_t)64 * KPAD * 2);      // final-token LN rows (padded)
  float*  logits = (float*)carve((size_t)16 * NCLSP * 4);
  (void)ws_size; (void)n_in; (void)in_sizes; (void)out_size;

  const dim3 B256(256);
  auto blocks1d = [](long total) { return dim3((unsigned)((total + 255) / 256)); };

  // embedding gather
  embed_kernel<<<blocks1d((long)M * DMODEL), B256, 0, stream>>>(ids, emb, x_f, M * DMODEL);

  // Wrel -> bf16 padded (80 x 96); only offsets 0..64 are ever read
  convert_pad_kernel<<<blocks1d((long)DVPAD * DQPAD), B256, 0, stream>>>(
      Wrel, Wrelb, DVPAD, DHEAD, DVPAD, DQPAD);

  for (int l = 0; l < LAYERS; l++) {
    const float* Wqkv_l = Wqkv + (size_t)l * D3 * DMODEL;
    const float* bqkv_l = bqkv + (size_t)l * D3;
    const float* W1_l   = W1 + (size_t)l * D4 * DMODEL;
    const float* b1_l   = b1 + (size_t)l * D4;
    const float* W2_l   = W2 + (size_t)l * DMODEL * D4;
    const float* b2_l   = b2 + (size_t)l * DMODEL;

    // ln1 -> hb (bf16, K padded)
    layernorm_bf16<<<dim3(M), B256, 0, stream>>>(x_f, ln1_w + (size_t)l * DMODEL,
                                                 ln1_b + (size_t)l * DMODEL, hb, 1, 0);
    // qkv = hb . Wqkv^T + bqkv  (bf16 out)
    convert_pad_kernel<<<blocks1d((long)D3 * KPAD), B256, 0, stream>>>(
        Wqkv_l, Wb, D3, DMODEL, D3, KPAD);
    gemm_bf16_wmma<<<dim3((D3 + 127) / 128, M / 64), dim3(128), 0, stream>>>(
        hb, Wb, bqkv_l, nullptr, qkvb, KPAD, D3, M, D3, D3, /*mode=*/1);

    // repack q/k (padded 96) and v (transposed, padded 80)
    repack_qkv_kernel<<<blocks1d((long)BATCH * NHEAD * SEQ * DQPAD), B256, 0, stream>>>(
        qkvb, qBb, kBb, vTb);

    // fused attention (+ residual into x)
    attention_kernel<<<dim3(BATCH * NHEAD * (SEQ / 16)), B256, SMEM_TOTAL, stream>>>(
        qBb, kBb, vTb, Wrelb, brel, x_f);

    // ln2 -> hb
    layernorm_bf16<<<dim3(M), B256, 0, stream>>>(x_f, ln2_w + (size_t)l * DMODEL,
                                                 ln2_b + (size_t)l * DMODEL, hb, 1, 0);
    // h1 = gelu(hb . W1^T + b1)  (bf16 out, fused GELU)
    convert_pad_kernel<<<blocks1d((long)D4 * KPAD), B256, 0, stream>>>(
        W1_l, Wb, D4, DMODEL, D4, KPAD);
    gemm_bf16_wmma<<<dim3((D4 + 127) / 128, M / 64), dim3(128), 0, stream>>>(
        hb, Wb, b1_l, nullptr, h1b, KPAD, D4, M, D4, D4, /*mode=*/3);

    // x += h1 . W2^T + b2  (residual f32 out)
    convert_pad_kernel<<<blocks1d((long)DMODEL * D4), B256, 0, stream>>>(
        W2_l, Wb, DMODEL, D4, DMODEL, D4);
    gemm_bf16_wmma<<<dim3((DMODEL + 127) / 128, M / 64), dim3(128), 0, stream>>>(
        h1b, Wb, b2_l, x_f, nullptr, D4, DMODEL, M, DMODEL, DMODEL, /*mode=*/4);
  }

  // final LN only on last token of each batch; pad M-tile with zeros
  hipMemsetAsync(headA, 0, (size_t)64 * KPAD * 2, stream);
  layernorm_bf16<<<dim3(BATCH), B256, 0, stream>>>(x_f, lnf_w, lnf_b, headA, SEQ, SEQ - 1);

  // head GEMM: logits = headA . Wh^T + bh
  convert_pad_kernel<<<blocks1d((long)NCLSP * KPAD), B256, 0, stream>>>(
      Wh, Wb, NCLS, DMODEL, NCLSP, KPAD);
  gemm_bf16_wmma<<<dim3((NCLSP + 127) / 128, 1), dim3(128), 0, stream>>>(
      headA, Wb, bh, logits, nullptr, KPAD, NCLS, 16, NCLSP, NCLSP, /*mode=*/0);

  // mask -> output
  mask_logits_kernel<<<blocks1d(BATCH * NCLS), B256, 0, stream>>>(logits, mask, out);
}
